// BiasedMultiheadAttention_69200513073396
// MI455X (gfx1250) — compile-verified, matrix operand order checked by reading, not run on hardware
//
#include <hip/hip_runtime.h>

// ---------------------------------------------------------------------------
// BiasedMultiheadAttention for MI455X (gfx1250, wave32, WMMA bf16 + TDM)
// B=2, L=2048, D=1024, H=16, d=64
// ---------------------------------------------------------------------------

#define B_  2
#define L_  2048
#define D_  1024
#define H_  16
#define HD_ 64
#define K3_ 3072          // 3*D
#define LARGE_NEG_ 10000.0f
#define CLAMP_ 20.0f
#define SM_SCALE_ 0.125f  // 1/sqrt(64)

typedef __attribute__((ext_vector_type(16))) __bf16 bf16x16;
typedef __attribute__((ext_vector_type(8)))  __bf16 bf16x8;
typedef __attribute__((ext_vector_type(8)))  float  f32x8;
typedef __attribute__((ext_vector_type(4)))  unsigned int u32x4;
typedef __attribute__((ext_vector_type(8)))  unsigned int u32x8;

// ---------------------------------------------------------------------------
// WMMA wrapper: D = A(16x32 bf16) * B(32x16 bf16) + C(16x16 f32)
// ---------------------------------------------------------------------------
__device__ __forceinline__ f32x8 wmma_bf16(bf16x16 a, bf16x16 b, f32x8 c) {
  return __builtin_amdgcn_wmma_f32_16x16x32_bf16(
      /*neg_a=*/false, a, /*neg_b=*/false, b,
      /*c_mod=*/(short)0, c, /*reuse_a=*/false, /*reuse_b=*/false);
}

// ---------------------------------------------------------------------------
// A/B fragment loader from a row-major [rows, ld] bf16 matrix (global or LDS;
// address-space inference turns the LDS case into ds_load_b128).
// A 16x32 layout (ISA 7.12.2): lanes 0-15 -> M=lane, K in {0..7, 16..23};
// lanes 16-31 -> M=lane-16, K in {8..15, 24..31}. Two 16B loads per lane.
// ---------------------------------------------------------------------------
__device__ __forceinline__ bf16x16 load_frag(const __bf16* tile, int ld, int lane) {
  const __bf16* p = tile + (size_t)(lane & 15) * (size_t)ld + ((lane & 16) >> 1);
  union { bf16x16 v; bf16x8 h[2]; } u;
  u.h[0] = *(const bf16x8*)(p);
  u.h[1] = *(const bf16x8*)(p + 16);
  return u.v;
}

// ---------------------------------------------------------------------------
// Tensor Data Mover: issue a 2-D tile load (tile_rows x 32 bf16 elements)
// from a row-major [nrows_total, row_len] bf16 tensor into LDS.
// D# built per CDNA5 ISA (2 SGPR groups, 2-D tensor).
// ---------------------------------------------------------------------------
__device__ __forceinline__ void tdm_load_tile_2d(unsigned lds_off,
                                                 const __bf16* gptr,
                                                 unsigned tile_rows,
                                                 unsigned row_len,
                                                 unsigned nrows_total) {
  unsigned long long ga = (unsigned long long)(uintptr_t)gptr;
  u32x4 g0;
  g0[0] = 1u;                                       // count=1, user mode, no gather
  g0[1] = lds_off;                                  // lds_addr (bytes)
  g0[2] = (unsigned)(ga & 0xFFFFFFFFu);             // global_addr[31:0]
  g0[3] = (unsigned)((ga >> 32) & 0x01FFFFFFu)      // global_addr[56:32]
          | (2u << 30);                             // type = 2 ("image")
  u32x8 g1;
  g1[0] = (1u << 16);                               // wg_mask=0, data_size=1 (2B)
  g1[1] = (row_len & 0xFFFFu) << 16;                // tensor_dim0[15:0]
  g1[2] = (row_len >> 16) | ((nrows_total & 0xFFFFu) << 16);  // dim0 hi | dim1 lo
  g1[3] = (nrows_total >> 16) | (32u << 16);        // dim1 hi | tile_dim0 = 32
  g1[4] = tile_rows;                                // tile_dim1 (tile_dim2 = 0)
  g1[5] = row_len;                                  // tensor_dim0_stride[31:0]
  g1[6] = 0u;                                       // stride0 hi | stride1 lo
  g1[7] = 0u;                                       // stride1 hi
  asm volatile("tensor_load_to_lds %0, %1" :: "s"(g0), "s"(g1) : "memory");
}

// ---------------------------------------------------------------------------
// f32 -> bf16 convert
// ---------------------------------------------------------------------------
__global__ void cvt_f32_bf16_kernel(const float* __restrict__ src,
                                    __bf16* __restrict__ dst, int n) {
  int i = blockIdx.x * blockDim.x + threadIdx.x;
  if (i < n) dst[i] = (__bf16)src[i];
}

// ---------------------------------------------------------------------------
// TDM-staged GEMM: C[M,N] = A[M,K=1024]bf16 @ W[N,K]bf16^T + bias.
// Block = 256 threads (8 waves), block tile 128x64, wave tile 32x32.
// Wave 0 drives double-buffered TDM panel loads (A: 128x32, B: 64x32) into
// LDS; all waves consume via ds_load_b128 fragments.
// QKV variant stores bf16 qkv (+ transposed V panel Vt[b,h,d,l]); OUT variant
// stores f32 + bias into d_out.
// ---------------------------------------------------------------------------
template <int NTOT, bool QKV>
__global__ __launch_bounds__(256) void gemm_tdm_kernel(
    const __bf16* __restrict__ A, const __bf16* __restrict__ Wb,
    const float* __restrict__ bias, __bf16* __restrict__ qkv,
    __bf16* __restrict__ Vt, float* __restrict__ outf, int mtot) {
  const int K = D_;
  __shared__ __bf16 lAs[2][128 * 32];
  __shared__ __bf16 lBs[2][64 * 32];

  int lane = threadIdx.x & 31, wave = threadIdx.x >> 5;
  int m0 = blockIdx.y * 128;
  int n0 = blockIdx.x * 64;
  int wm = (wave >> 1) * 32;  // wave row offset within 128-row panel
  int wn = (wave & 1) * 32;   // wave col offset within 64-col panel

  bool issuer = (wave == 0);
  if (issuer) {
    tdm_load_tile_2d((unsigned)(uintptr_t)&lAs[0][0], A + (size_t)m0 * K, 128,
                     K, (unsigned)mtot);
    tdm_load_tile_2d((unsigned)(uintptr_t)&lBs[0][0], Wb + (size_t)n0 * K, 64,
                     K, (unsigned)NTOT);
  }

  f32x8 acc00 = {}, acc01 = {}, acc10 = {}, acc11 = {};
  const int NSTEP = K / 32;
  for (int i = 0; i < NSTEP; i++) {
    int cur = i & 1;
    if (issuer) __builtin_amdgcn_s_wait_tensorcnt(0);
    __syncthreads();  // buf[cur] ready; prior reads of buf[1-cur] retired
    if (issuer && (i + 1) < NSTEP) {
      int kk = (i + 1) * 32;
      tdm_load_tile_2d((unsigned)(uintptr_t)&lAs[1 - cur][0],
                       A + (size_t)m0 * K + kk, 128, K, (unsigned)mtot);
      tdm_load_tile_2d((unsigned)(uintptr_t)&lBs[1 - cur][0],
                       Wb + (size_t)n0 * K + kk, 64, K, (unsigned)NTOT);
    }
    const __bf16* la = &lAs[cur][0];
    const __bf16* lb = &lBs[cur][0];
    bf16x16 a0 = load_frag(la + (wm + 0) * 32, 32, lane);
    bf16x16 a1 = load_frag(la + (wm + 16) * 32, 32, lane);
    bf16x16 b0 = load_frag(lb + (wn + 0) * 32, 32, lane);
    bf16x16 b1 = load_frag(lb + (wn + 16) * 32, 32, lane);
    acc00 = wmma_bf16(a0, b0, acc00);
    acc01 = wmma_bf16(a0, b1, acc01);
    acc10 = wmma_bf16(a1, b0, acc10);
    acc11 = wmma_bf16(a1, b1, acc11);
  }

  int rowoff = (lane & 16) ? 8 : 0;
  int cn = lane & 15;
  f32x8* accs[2][2] = {{&acc00, &acc01}, {&acc10, &acc11}};
#pragma unroll
  for (int i = 0; i < 2; i++) {
#pragma unroll
    for (int j = 0; j < 2; j++) {
      f32x8 a = *accs[i][j];
#pragma unroll
      for (int r = 0; r < 8; r++) {
        int row = m0 + wm + 16 * i + r + rowoff;
        int col = n0 + wn + 16 * j + cn;
        float v = a[r] + bias[col];
        if (QKV) {
          __bf16 bv = (__bf16)v;
          qkv[(size_t)row * NTOT + col] = bv;
          if (col >= 2 * D_) {  // V -> transposed per-head layout Vt[b,h,d,l]
            int bb = row >> 11;
            int lpos = row & (L_ - 1);
            int hv = (col - 2 * D_) >> 6;
            int dd = (col - 2 * D_) & 63;
            Vt[(size_t)((bb * H_ + hv) * HD_ + dd) * L_ + lpos] = bv;
          }
        } else {
          outf[(size_t)row * NTOT + col] = v;
        }
      }
    }
  }
}

// ---------------------------------------------------------------------------
// Flash attention: one wave owns 16 queries x full head-dim (64).
// 32 keys per iteration: 4 WMMAs (QK^T, K-dim=64 split in 2) + 4 WMMAs (P.V).
// Online softmax with running max/sum; clamp(-20) matches reference semantics.
// attn_bias streamed with non-temporal loads (read-once, 536 MB).
// ---------------------------------------------------------------------------
__global__ __launch_bounds__(256) void attn_kernel(
    const __bf16* __restrict__ qkv, const __bf16* __restrict__ Vt,
    const float* __restrict__ bias, const int* __restrict__ mask,
    __bf16* __restrict__ att) {
  __shared__ __bf16 plds[8][16 * 32];  // per-wave P-tile bounce buffer

  int lane = threadIdx.x & 31, wave = threadIdx.x >> 5;
  int id = blockIdx.x * 8 + wave;  // 0..4095
  int qt = id & 127;
  int bh = id >> 7;
  int h = bh & (H_ - 1);
  int b = bh >> 4;
  int q0 = qt << 4;

  const __bf16* qbase = qkv + (size_t)(b * L_ + q0) * K3_ + h * HD_;
  bf16x16 qa0 = load_frag(qbase + 0, K3_, lane);
  bf16x16 qa1 = load_frag(qbase + 32, K3_, lane);

  f32x8 o0 = {}, o1 = {}, o2 = {}, o3 = {};
  float m[8], l[8];
#pragma unroll
  for (int r = 0; r < 8; r++) { m[r] = -3.0e38f; l[r] = 0.0f; }

  int rowoff = (lane & 16) ? 8 : 0;
  int cn = lane & 15;
  const float* bias_base = bias + ((size_t)(b * H_ + h) * L_ + q0) * L_;
  const int* mask_base = mask + b * L_;
  const __bf16* kbase = qkv + (size_t)b * L_ * K3_ + D_ + h * HD_;
  const __bf16* vbase = Vt + (size_t)((b * H_ + h) * HD_) * L_;
  __bf16* myp = &plds[wave][0];

  for (int kt = 0; kt < L_; kt += 32) {
    // ---- S = Q K^T (two 16-key column tiles) ----
    bf16x16 kb00 = load_frag(kbase + (size_t)kt * K3_ + 0, K3_, lane);
    bf16x16 kb01 = load_frag(kbase + (size_t)kt * K3_ + 32, K3_, lane);
    bf16x16 kb10 = load_frag(kbase + (size_t)(kt + 16) * K3_ + 0, K3_, lane);
    bf16x16 kb11 = load_frag(kbase + (size_t)(kt + 16) * K3_ + 32, K3_, lane);
    f32x8 s0 = {}, s1 = {};
    s0 = wmma_bf16(qa0, kb00, s0);
    s0 = wmma_bf16(qa1, kb01, s0);
    s1 = wmma_bf16(qa0, kb10, s1);
    s1 = wmma_bf16(qa1, kb11, s1);

    float pad0 = LARGE_NEG_ * (float)mask_base[kt + cn];
    float pad1 = LARGE_NEG_ * (float)mask_base[kt + 16 + cn];

    float sv0[8], sv1[8];
#pragma unroll
    for (int r = 0; r < 8; r++) {
      float bb0 = __builtin_nontemporal_load(bias_base + (size_t)(r + rowoff) * L_ + kt + cn);
      float bb1 = __builtin_nontemporal_load(bias_base + (size_t)(r + rowoff) * L_ + kt + 16 + cn);
      sv0[r] = s0[r] * SM_SCALE_ + bb0 - pad0;
      sv1[r] = s1[r] * SM_SCALE_ + bb1 - pad1;
    }

    float corr[8];
#pragma unroll
    for (int r = 0; r < 8; r++) {
      // row max across the 16 lanes holding this row (C layout)
      float tm = fmaxf(sv0[r], sv1[r]);
      tm = fmaxf(tm, __shfl_xor(tm, 1, 32));
      tm = fmaxf(tm, __shfl_xor(tm, 2, 32));
      tm = fmaxf(tm, __shfl_xor(tm, 4, 32));
      tm = fmaxf(tm, __shfl_xor(tm, 8, 32));
      float nm = fmaxf(m[r], tm);
      corr[r] = __expf(m[r] - nm);
      m[r] = nm;
      sv0[r] = __expf(fmaxf(sv0[r] - nm, -CLAMP_));
      sv1[r] = __expf(fmaxf(sv1[r] - nm, -CLAMP_));
      float rs = sv0[r] + sv1[r];
      rs += __shfl_xor(rs, 1, 32);
      rs += __shfl_xor(rs, 2, 32);
      rs += __shfl_xor(rs, 4, 32);
      rs += __shfl_xor(rs, 8, 32);
      l[r] = l[r] * corr[r] + rs;
    }
#pragma unroll
    for (int r = 0; r < 8; r++) {
      o0[r] *= corr[r]; o1[r] *= corr[r]; o2[r] *= corr[r]; o3[r] *= corr[r];
    }

    // ---- C-layout P tile -> A-fragment layout via wave-private LDS ----
#pragma unroll
    for (int r = 0; r < 8; r++) {
      myp[(r + rowoff) * 32 + cn] = (__bf16)sv0[r];
      myp[(r + rowoff) * 32 + 16 + cn] = (__bf16)sv1[r];
    }
    asm volatile("s_wait_dscnt 0" ::: "memory");
    bf16x16 pa = load_frag(myp, 32, lane);

    // ---- O += P V  (Vt rows are d-indices -> contiguous key runs) ----
    bf16x16 vb0 = load_frag(vbase + (size_t)(0 * 16) * L_ + kt, L_, lane);
    bf16x16 vb1 = load_frag(vbase + (size_t)(1 * 16) * L_ + kt, L_, lane);
    bf16x16 vb2 = load_frag(vbase + (size_t)(2 * 16) * L_ + kt, L_, lane);
    bf16x16 vb3 = load_frag(vbase + (size_t)(3 * 16) * L_ + kt, L_, lane);
    o0 = wmma_bf16(pa, vb0, o0);
    o1 = wmma_bf16(pa, vb1, o1);
    o2 = wmma_bf16(pa, vb2, o2);
    o3 = wmma_bf16(pa, vb3, o3);
  }

  // ---- normalize and store attention output (bf16, [B,L,D] layout) ----
#pragma unroll
  for (int r = 0; r < 8; r++) {
    float inv = 1.0f / l[r];
    int row = b * L_ + q0 + r + rowoff;
    __bf16* arow = att + (size_t)row * D_ + h * HD_ + cn;
    arow[0]  = (__bf16)(o0[r] * inv);
    arow[16] = (__bf16)(o1[r] * inv);
    arow[32] = (__bf16)(o2[r] * inv);
    arow[48] = (__bf16)(o3[r] * inv);
  }
}

// ---------------------------------------------------------------------------
// Launch
// ---------------------------------------------------------------------------
extern "C" void kernel_launch(void* const* d_in, const int* in_sizes, int n_in,
                              void* d_out, int out_size, void* d_ws, size_t ws_size,
                              hipStream_t stream) {
  const float* x     = (const float*)d_in[0];  // [B,L,D]
  const int*   kmask = (const int*)d_in[1];    // [B,L]
  const float* abias = (const float*)d_in[2];  // [B,H,L,L]
  const float* W_in  = (const float*)d_in[3];  // [3D,D]
  const float* b_in  = (const float*)d_in[4];  // [3D]
  const float* W_out = (const float*)d_in[5];  // [D,D]
  const float* b_out = (const float*)d_in[6];  // [D]
  float* out = (float*)d_out;                  // [B,L,D]

  const size_t ML = (size_t)B_ * L_;  // 4096
  char* ws = (char*)d_ws;
  auto align256 = [](size_t v) { return (v + 255) & ~(size_t)255; };

  size_t off = 0;
  __bf16* xbf = (__bf16*)(ws + off);    off = align256(off + ML * D_ * sizeof(__bf16));
  __bf16* winbf = (__bf16*)(ws + off);  off = align256(off + (size_t)K3_ * D_ * sizeof(__bf16));
  __bf16* woutbf = (__bf16*)(ws + off); off = align256(off + (size_t)D_ * D_ * sizeof(__bf16));
  __bf16* qkvbf = (__bf16*)(ws + off);  off = align256(off + ML * K3_ * sizeof(__bf16));
  __bf16* vt = (__bf16*)(ws + off);     off = align256(off + (size_t)B_ * H_ * HD_ * L_ * sizeof(__bf16));
  __bf16* att = (__bf16*)(ws + off);    off = align256(off + ML * D_ * sizeof(__bf16));
  (void)ws_size; (void)in_sizes; (void)n_in; (void)out_size;

  // 1) bf16 conversions
  {
    int n1 = (int)(ML * D_);
    cvt_f32_bf16_kernel<<<(n1 + 255) / 256, 256, 0, stream>>>(x, xbf, n1);
    int n2 = K3_ * D_;
    cvt_f32_bf16_kernel<<<(n2 + 255) / 256, 256, 0, stream>>>(W_in, winbf, n2);
    int n3 = D_ * D_;
    cvt_f32_bf16_kernel<<<(n3 + 255) / 256, 256, 0, stream>>>(W_out, woutbf, n3);
  }

  // 2) QKV projection (M=4096, N=3072, K=1024), TDM-staged; also emits Vt
  gemm_tdm_kernel<K3_, true><<<dim3(K3_ / 64, (unsigned)(ML / 128)), 256, 0, stream>>>(
      xbf, winbf, b_in, qkvbf, vt, nullptr, (int)ML);

  // 3) Flash attention: 4096 query tiles, 8 waves per block
  attn_kernel<<<(B_ * H_ * (L_ / 16)) / 8, 256, 0, stream>>>(
      qkvbf, vt, abias, kmask, att);

  // 4) Output projection (M=4096, N=1024, K=1024), TDM-staged -> f32 d_out
  gemm_tdm_kernel<D_, false><<<dim3(D_ / 64, (unsigned)(ML / 128)), 256, 0, stream>>>(
      att, woutbf, b_out, nullptr, nullptr, out, (int)ML);
}
